// CodecAttention_88948772700260
// MI455X (gfx1250) — matrix operand order repaired and checked by reading, not compiled, not run on hardware
//
#include <hip/hip_runtime.h>
#include <hip/hip_bf16.h>

typedef __attribute__((ext_vector_type(16))) _Float16 v16h;
typedef __attribute__((ext_vector_type(8)))  float    v8f;

#define S_LEN  2048
#define BATCH  2
#define MODEL  1024
#define HEADS  16
#define KVH    4
#define HD     64
#define KVD    (KVH * HD)         // 256
#define WIN    512
#define ROWS   (BATCH * S_LEN)    // 4096

// ---------------------------------------------------------------------------
// Bulk f32 -> f16 conversion (hoists all cvt work out of the WMMA hot loops).
// ---------------------------------------------------------------------------
__global__ __launch_bounds__(256)
void f32_to_f16(const float* __restrict__ in, _Float16* __restrict__ out,
                int n) {
  const int i = blockIdx.x * 256 + (int)threadIdx.x;
  if (i < n) out[i] = (_Float16)in[i];
}

// ---------------------------------------------------------------------------
// C[M,N] = A[M,K=1024] @ W[N,K=1024]^T, f16 operands, f32 accumulate/output.
// Block = 128 threads (4 waves). Each wave owns a 32(M) x 64(N) strip:
// 8 accumulator tiles, 2 A fragments per K-step, every B fragment reused by
// two WMMAs (8 wmma : 12 b128-loads per K=32 step).
// Fragment layouts per CDNA5 ISA §7.12.2:
//   A 16x32 f16 : lane m=L%16; lanes<16 hold K 0..7 & 16..23, lanes>=16 hold
//                 K 8..15 & 24..31.
//   B 32x16 f16 : lane n=L%16; lanes<16 hold K 0..7 & 16..23, lanes>=16 hold
//                 K 8..15 & 24..31.
//   C 16x16 f32 : vgpr r, lane L -> (M = r + (L>=16 ? 8 : 0), N = L%16).
// ---------------------------------------------------------------------------
__global__ __launch_bounds__(128)
void gemm_h(const _Float16* __restrict__ A, const _Float16* __restrict__ W,
            float* __restrict__ C, int N) {
  const int K = MODEL;
  const unsigned lane = threadIdx.x & 31u;
  const unsigned wave = threadIdx.x >> 5;
  const unsigned l16  = lane & 15u;
  const bool     hi   = lane >= 16;
  const int m0 = blockIdx.y * 128 + (int)wave * 32;
  const int n0 = blockIdx.x * 64;

  v8f acc[2][4] = {};
  const _Float16* arow0 = A + (size_t)(m0 + (int)l16) * K;
  const _Float16* arow1 = arow0 + (size_t)16 * K;
  const int kbA = hi ? 8 : 0;
  const int kbB = hi ? 16 : 0;

  for (int k0 = 0; k0 < K; k0 += 32) {
    __builtin_prefetch((const void*)(arow0 + k0 + 128), 0, 1);  // global_prefetch
    v16h a0, a1;
#pragma unroll
    for (int j = 0; j < 8; ++j) {
      a0[j]     = arow0[k0 + kbA + j];
      a0[j + 8] = arow0[k0 + kbA + 16 + j];
      a1[j]     = arow1[k0 + kbA + j];
      a1[j + 8] = arow1[k0 + kbA + 16 + j];
    }
#pragma unroll
    for (int c = 0; c < 4; ++c) {
      const _Float16* wrow = W + (size_t)(n0 + c * 16 + (int)l16) * K;
      v16h bf;
#pragma unroll
      for (int j = 0; j < 16; ++j) bf[j] = wrow[k0 + kbA + (j & 7) + ((j >> 3) << 4)];
      acc[0][c] = __builtin_amdgcn_wmma_f32_16x16x32_f16(
          false, a0, false, bf, (short)0, acc[0][c], false, false);
      acc[1][c] = __builtin_amdgcn_wmma_f32_16x16x32_f16(
          false, a1, false, bf, (short)0, acc[1][c], false, false);
    }
  }

#pragma unroll
  for (int halfm = 0; halfm < 2; ++halfm) {
    const int mb = m0 + halfm * 16 + (hi ? 8 : 0);
#pragma unroll
    for (int c = 0; c < 4; ++c)
#pragma unroll
      for (int r = 0; r < 8; ++r)
        C[(size_t)(mb + r) * N + n0 + c * 16 + (int)l16] = acc[halfm][c][r];
  }
}

// ---------------------------------------------------------------------------
// Row-wise RMSNorm: read f32 rows (full precision reduction), write f16 rows.
// ---------------------------------------------------------------------------
__global__ __launch_bounds__(256)
void rmsnorm_f16(const float* __restrict__ in, const float* __restrict__ w,
                 _Float16* __restrict__ out, int width) {
  const float* p = in + (size_t)blockIdx.x * width;
  _Float16* q = out + (size_t)blockIdx.x * width;
  float ss = 0.f;
  for (int i = threadIdx.x; i < width; i += 256) {
    float x = p[i];
    ss += x * x;
  }
  __shared__ float red[256];
  red[threadIdx.x] = ss;
  __syncthreads();
  for (int s = 128; s > 0; s >>= 1) {
    if ((int)threadIdx.x < s) red[threadIdx.x] += red[threadIdx.x + s];
    __syncthreads();
  }
  const float inv = rsqrtf(red[0] / (float)width + 1e-6f);
  for (int i = threadIdx.x; i < width; i += 256)
    q[i] = (_Float16)(p[i] * inv * w[i]);
}

// ---------------------------------------------------------------------------
// Sliding-window GQA attention with ALiBi, online softmax. One wave per
// (batch, head, 16-query tile); 32 keys per iteration:
//   scores : 2 tiles x (2 x v_wmma K=32)  = 4 WMMAs
//   PV     : full 16x32 P A-fragment, 4 x v_wmma = 4 WMMAs
// Key range is front-aligned to 32 (may start below 0 / window edge); such
// keys are masked (key<0 added to mask) and their loads index-clamped to 0.
// ---------------------------------------------------------------------------
__global__ __launch_bounds__(32)
void attn_kernel(const _Float16* __restrict__ q, const _Float16* __restrict__ k,
                 const _Float16* __restrict__ v, _Float16* __restrict__ y) {
  const unsigned lane = threadIdx.x & 31u;
  const unsigned l16  = lane & 15u;
  const bool     hi   = lane >= 16;
  const int wg  = blockIdx.x;           // B*H*(S/16) = 4096
  const int qt  = wg & 127;             // S/16 = 128
  const int h   = (wg >> 7) & 15;
  const int b   = wg >> 11;
  const int q0  = qt * 16;
  const int kvh = h >> 2;               // GQA: 4 q-heads per kv-head
  const float slope = exp2f(-0.5f * (float)(h + 1));  // ALiBi, 16 heads
  const float scale = 0.125f;                          // 1/sqrt(64)

  // Q tile 16x64 -> two 16x32 f16 A fragments (loaded once).
  const _Float16* qrow =
      q + (size_t)(b * S_LEN + q0 + (int)l16) * (HEADS * HD) + h * HD;
  const int kbA = hi ? 8 : 0;
  v16h qa[2];
#pragma unroll
  for (int c2 = 0; c2 < 2; ++c2)
#pragma unroll
    for (int j = 0; j < 8; ++j) {
      qa[c2][j]     = qrow[c2 * 32 + kbA + j];
      qa[c2][j + 8] = qrow[c2 * 32 + kbA + 16 + j];
    }

  const float NEG_INF = -__builtin_inff();
  float mrow[8], lrow[8];
  v8f o[4] = {};
#pragma unroll
  for (int r = 0; r < 8; ++r) { mrow[r] = NEG_INF; lrow[r] = 0.f; }

  __shared__ _Float16 pbuf[16][32];

  const int mb  = hi ? 8 : 0;
  const int kbB = hi ? 16 : 0;
  const int kst  = (q0 >= WIN) ? (q0 - WIN) : 0;   // first legal key
  const int kend = q0 + 16;                        // last key + 1
  const int nIter = (kend - kst + 31) / 32;
  const int start = kend - nIter * 32;             // may be < 0 (masked)

  for (int it = 0; it < nIter; ++it) {
    const int kt = start + it * 32;

    // ---- scores: two 16x16 tiles over keys [kt, kt+32)
    v8f s[2] = {};
#pragma unroll
    for (int t = 0; t < 2; ++t) {
      int kr = kt + t * 16 + (int)l16;
      kr = kr < 0 ? 0 : kr;                        // clamp (masked anyway)
#pragma unroll
      for (int c2 = 0; c2 < 2; ++c2) {
        const _Float16* krow =
            k + (size_t)(b * S_LEN + kr) * KVD + kvh * HD + c2 * 32 + kbA;
        v16h bf;
#pragma unroll
        for (int j = 0; j < 16; ++j) bf[j] = krow[(j & 7) + ((j >> 3) << 4)];
        s[t] = __builtin_amdgcn_wmma_f32_16x16x32_f16(
            false, qa[c2], false, bf, (short)0, s[t], false, false);
      }
    }

    __syncthreads();  // WAR guard on pbuf vs previous iteration's reads
    // ---- bias + mask + online softmax (row stats over 16-lane half-group)
#pragma unroll
    for (int r = 0; r < 8; ++r) {
      const int m    = mb + r;
      const int qi   = q0 + m;
      const int key0 = kt + (int)l16;
      const int key1 = key0 + 16;
      const int rel0 = key0 - qi;
      const int rel1 = key1 - qi;
      float x0 = (rel0 > 0 || rel0 < -WIN || key0 < 0)
                     ? NEG_INF : s[0][r] * scale + slope * (float)rel0;
      float x1 = (rel1 > 0 || rel1 < -WIN || key1 < 0)
                     ? NEG_INF : s[1][r] * scale + slope * (float)rel1;
      float tmax = fmaxf(x0, x1);
      tmax = fmaxf(tmax, __shfl_xor(tmax, 1));
      tmax = fmaxf(tmax, __shfl_xor(tmax, 2));
      tmax = fmaxf(tmax, __shfl_xor(tmax, 4));
      tmax = fmaxf(tmax, __shfl_xor(tmax, 8));
      const float newm = fmaxf(mrow[r], tmax);
      const float meff = (newm == NEG_INF) ? 0.f : newm;  // NaN guard
      const float alpha = __expf(mrow[r] - meff);
      const float p0    = __expf(x0 - meff);
      const float p1    = __expf(x1 - meff);
      float psum = p0 + p1;
      psum += __shfl_xor(psum, 1);
      psum += __shfl_xor(psum, 2);
      psum += __shfl_xor(psum, 4);
      psum += __shfl_xor(psum, 8);
      mrow[r] = newm;
      lrow[r] = lrow[r] * alpha + psum;
#pragma unroll
      for (int c = 0; c < 4; ++c) o[c][r] *= alpha;
      pbuf[m][l16]      = (_Float16)p0;
      pbuf[m][l16 + 16] = (_Float16)p1;
    }
    __syncthreads();

    // ---- P A-fragment: full 16x32, no padding
    v16h pa;
    {
      const int kbP = hi ? 8 : 0;
#pragma unroll
      for (int j = 0; j < 8; ++j) {
        pa[j]     = pbuf[l16][kbP + j];
        pa[j + 8] = pbuf[l16][16 + kbP + j];
      }
    }
    // ---- O += P(16x32) . V(32x64): 4 WMMAs across head-dim chunks
#pragma unroll
    for (int c = 0; c < 4; ++c) {
      v16h vb;
#pragma unroll
      for (int j = 0; j < 16; ++j) {
        int vr = kt + kbA + (j & 7) + ((j >> 3) << 4);
        vr = vr < 0 ? 0 : vr;                      // clamp (p == 0 there)
        vb[j] = v[(size_t)(b * S_LEN + vr) * KVD + kvh * HD + c * 16 +
                  (int)l16];
      }
      o[c] = __builtin_amdgcn_wmma_f32_16x16x32_f16(
          false, pa, false, vb, (short)0, o[c], false, false);
    }
  }

  // ---- normalize + write y (f16) for the output projection
#pragma unroll
  for (int c = 0; c < 4; ++c)
#pragma unroll
    for (int r = 0; r < 8; ++r) {
      const int m = mb + r;
      y[(size_t)(b * S_LEN + q0 + m) * (HEADS * HD) + h * HD + c * 16 +
        (int)l16] = (_Float16)(o[c][r] / lrow[r]);
    }
}

// ---------------------------------------------------------------------------
extern "C" void kernel_launch(void* const* d_in, const int* in_sizes, int n_in,
                              void* d_out, int out_size, void* d_ws,
                              size_t ws_size, hipStream_t stream) {
  (void)in_sizes; (void)n_in; (void)out_size; (void)ws_size;
  const float* x  = (const float*)d_in[0];
  const float* wq = (const float*)d_in[1];
  const float* wk = (const float*)d_in[2];
  const float* wv = (const float*)d_in[3];
  const float* wo = (const float*)d_in[4];
  const float* qw = (const float*)d_in[5];
  const float* kw = (const float*)d_in[6];
  float* out = (float*)d_out;

  // ---- workspace carve-out (~60 MB)
  char* w = (char*)d_ws;
  float* qraw = (float*)w;  w += (size_t)ROWS * MODEL * 4;   // raw x@wq^T
  float* kraw = (float*)w;  w += (size_t)ROWS * KVD * 4;     // raw x@wk^T
  float* vraw = (float*)w;  w += (size_t)ROWS * KVD * 4;     // raw x@wv^T
  _Float16* xh  = (_Float16*)w; w += (size_t)ROWS * MODEL * 2;
  _Float16* wqh = (_Float16*)w; w += (size_t)MODEL * MODEL * 2;
  _Float16* wkh = (_Float16*)w; w += (size_t)KVD * MODEL * 2;
  _Float16* wvh = (_Float16*)w; w += (size_t)KVD * MODEL * 2;
  _Float16* woh = (_Float16*)w; w += (size_t)MODEL * MODEL * 2;
  _Float16* qh  = (_Float16*)w; w += (size_t)ROWS * MODEL * 2;
  _Float16* kh  = (_Float16*)w; w += (size_t)ROWS * KVD * 2;
  _Float16* vh  = (_Float16*)w; w += (size_t)ROWS * KVD * 2;
  _Float16* yh  = (_Float16*)w; w += (size_t)ROWS * MODEL * 2;

  // ---- one-time operand conversions (bandwidth-trivial)
  f32_to_f16<<<(ROWS * MODEL) / 256, 256, 0, stream>>>(x, xh, ROWS * MODEL);
  f32_to_f16<<<(MODEL * MODEL) / 256, 256, 0, stream>>>(wq, wqh, MODEL * MODEL);
  f32_to_f16<<<(KVD * MODEL) / 256, 256, 0, stream>>>(wk, wkh, KVD * MODEL);
  f32_to_f16<<<(KVD * MODEL) / 256, 256, 0, stream>>>(wv, wvh, KVD * MODEL);
  f32_to_f16<<<(MODEL * MODEL) / 256, 256, 0, stream>>>(wo, woh, MODEL * MODEL);

  // ---- QKV projections (f16 WMMA, f32 out)
  const dim3 gblk(128);
  gemm_h<<<dim3(MODEL / 64, ROWS / 128), gblk, 0, stream>>>(xh, wqh, qraw, MODEL);
  gemm_h<<<dim3(KVD / 64, ROWS / 128), gblk, 0, stream>>>(xh, wkh, kraw, KVD);
  gemm_h<<<dim3(KVD / 64, ROWS / 128), gblk, 0, stream>>>(xh, wvh, vraw, KVD);

  // ---- norms (f32 reduce) -> f16 activations
  rmsnorm_f16<<<ROWS, 256, 0, stream>>>(qraw, qw, qh, MODEL);
  rmsnorm_f16<<<ROWS, 256, 0, stream>>>(kraw, kw, kh, KVD);
  f32_to_f16<<<(ROWS * KVD) / 256, 256, 0, stream>>>(vraw, vh, ROWS * KVD);

  // ---- attention
  attn_kernel<<<BATCH * HEADS * (S_LEN / 16), 32, 0, stream>>>(qh, kh, vh, yh);

  // ---- output projection
  gemm_h<<<dim3(MODEL / 64, ROWS / 128), gblk, 0, stream>>>(yh, woh, out, MODEL);
}
